// TransformerPredictor_25262997635279
// MI455X (gfx1250) — matrix operand order repaired and checked by reading, not compiled, not run on hardware
//
#include <hip/hip_runtime.h>

typedef unsigned short u16;
typedef unsigned int u32;
typedef unsigned long long u64;
typedef __attribute__((ext_vector_type(16))) __bf16 v16bf;
typedef __attribute__((ext_vector_type(8)))  float  v8f;
typedef __attribute__((ext_vector_type(4)))  u32    v4u;
typedef __attribute__((ext_vector_type(8)))  int    v8i;
typedef __attribute__((ext_vector_type(4)))  int    v4i;

// ---------------------------------------------------------------------------
// Problem constants
// ---------------------------------------------------------------------------
#define N_OBJ   8192
#define N_REL   65536
#define HID     512
#define POOL    4096
#define NOC     151
#define NRC     51
#define NPAD    64          // NRC padded to 64 for WMMA tiles

#if __has_builtin(__builtin_amdgcn_tensor_load_to_lds)
#define HAVE_TDM 1
#else
#define HAVE_TDM 0
#endif

// ---------------------------------------------------------------------------
// Helpers
// ---------------------------------------------------------------------------
union Frag {                 // one WMMA A/B operand: 8 VGPRs / 32 bytes per lane
  uint4 q[2];
  v16bf bf;
};

__device__ __forceinline__ u16 f2bf(float f) {
  u32 u = __builtin_bit_cast(u32, f);
  u += 0x7fffu + ((u >> 16) & 1u);         // round-to-nearest-even
  return (u16)(u >> 16);
}

__device__ __forceinline__ v8f wmma_bf16(const Frag& a, const Frag& b, v8f c) {
  // v_wmma_f32_16x16x32_bf16  D = A*B + C   (wave32, f32 accumulate)
  return __builtin_amdgcn_wmma_f32_16x16x32_bf16(
      /*neg_a=*/false, a.bf, /*neg_b=*/false, b.bf,
      /*c_mod=*/(short)0, c, /*reuse_a=*/false, /*reuse_b=*/false);
}

__device__ __forceinline__ void wait_tensor0() {
#if __has_builtin(__builtin_amdgcn_s_wait_tensorcnt)
  __builtin_amdgcn_s_wait_tensorcnt(0);
#else
  asm volatile("s_wait_tensorcnt 0x0" ::: "memory");
#endif
}

#if HAVE_TDM
// Issue one TDM 2-D tile load: global (bf16, row stride `stride0` elements)
// -> LDS at byte offset `lds_off`. Descriptor per CDNA5 ISA ch.8 (D# groups).
__device__ __forceinline__ void tdm_load_2d(u32 lds_off, const void* gptr,
                                            u32 td0, u32 td1,
                                            u32 tile0, u32 tile1,
                                            u32 stride0) {
  u64 ga = (u64)gptr;
  v4u g0;
  g0[0] = 1u;                                        // count=1 (valid), user mode
  g0[1] = lds_off;                                   // lds_addr (bytes)
  g0[2] = (u32)(ga & 0xffffffffu);                   // global_addr[31:0]
  g0[3] = (u32)((ga >> 32) & 0x01ffffffu) | (2u << 30); // addr[56:32] | type=2
  v8i g1;
  g1[0] = (int)(1u << 16);                           // data_size=1 -> 2 bytes
  g1[1] = (int)((td0 & 0xffffu) << 16);              // tensor_dim0[15:0]
  g1[2] = (int)((td0 >> 16) | ((td1 & 0xffffu) << 16));
  g1[3] = (int)((td1 >> 16) | (tile0 << 16));        // tile_dim0
  g1[4] = (int)tile1;                                // tile_dim1 (tile_dim2=0)
  g1[5] = (int)stride0;                              // tensor_dim0_stride[31:0]
  g1[6] = 0;
  g1[7] = 0;
  v4i z4 = {0, 0, 0, 0};
#if defined(__clang_major__) && __clang_major__ >= 23
  v8i z8 = {0, 0, 0, 0, 0, 0, 0, 0};
  __builtin_amdgcn_tensor_load_to_lds(g0, g1, z4, z4, z8, 0);
#else
  __builtin_amdgcn_tensor_load_to_lds(g0, g1, z4, z4, 0);
#endif
}
#endif

// ---------------------------------------------------------------------------
// Kernel 1: straight f32 -> bf16 conversion (edge_ctx)
// ---------------------------------------------------------------------------
__global__ void k_cvt(const float* __restrict__ src, u16* __restrict__ dst, int n) {
  int i = blockIdx.x * 256 + threadIdx.x;
  if (i < n) dst[i] = f2bf(src[i]);
}

// ---------------------------------------------------------------------------
// Kernel 2: transpose+convert  src f32 (K x Nsrc) row-major -> dst bf16 (Ndst x K)
// rows n >= Nsrc are zero padding (used to pad 51 -> 64 classes)
// ---------------------------------------------------------------------------
__global__ void k_tcvt(const float* __restrict__ src, u16* __restrict__ dst,
                       int K, int Nsrc, int Ndst) {
  int i = blockIdx.x * 256 + threadIdx.x;
  int total = Ndst * K;
  if (i >= total) return;
  int n = i / K;
  int k = i - n * K;
  dst[i] = (n < Nsrc) ? f2bf(src[(size_t)k * Nsrc + n]) : (u16)0;
}

// ---------------------------------------------------------------------------
// Kernel 3: GEMM1  edge_rep = edge_ctx @ W_post_emb + b   (8192x1024), bf16 out
// One wave per 16x64 output strip; K=512 in 16 WMMA steps. A/B from global
// (both operands small and L2-resident: 8MB + 1MB).
// ---------------------------------------------------------------------------
__global__ __launch_bounds__(256)
void k_gemm1(const u16* __restrict__ ecb,     // 8192x512 bf16 row-major
             const u16* __restrict__ wpeT,    // 1024x512 bf16 (transposed W)
             const float* __restrict__ bpe,   // 1024
             u16* __restrict__ erep) {        // 8192x1024 bf16
  int w  = threadIdx.x >> 5;
  int ln = threadIdx.x & 31;
  int lh = ln >> 4, lc = ln & 15;
  int gw = blockIdx.x * 8 + w;               // 8192 waves total
  int mt = gw >> 4;                          // 0..511  (16-row tile)
  int ng = gw & 15;                          // 0..15   (64-col strip)
  int row = mt * 16 + lc;

  v8f g[4] = {};
  for (int k = 0; k < HID; k += 32) {
    Frag a;
    const uint4* ap = (const uint4*)(ecb + (size_t)row * HID + k + 16 * lh);
    a.q[0] = ap[0];
    a.q[1] = ap[1];                          // +8 elements
#pragma unroll
    for (int nt = 0; nt < 4; ++nt) {
      int col = ng * 64 + nt * 16 + lc;
      Frag b;
      const uint4* bp = (const uint4*)(wpeT + (size_t)col * HID + k + 16 * lh);
      b.q[0] = bp[0];
      b.q[1] = bp[1];
      g[nt] = wmma_bf16(a, b, g[nt]);
    }
  }
#pragma unroll
  for (int nt = 0; nt < 4; ++nt) {
    int col = ng * 64 + nt * 16 + lc;
    float bias = bpe[col];
#pragma unroll
    for (int j = 0; j < 8; ++j) {
      int r2 = mt * 16 + j + 8 * lh;
      erep[(size_t)r2 * 1024 + col] = f2bf(g[nt][j] + bias);
    }
  }
}

// ---------------------------------------------------------------------------
// Kernel 4: gather prod_rep rows: [head(i0) | tail(i1)]  (bf16)
// ---------------------------------------------------------------------------
__global__ void k_gather(const u16* __restrict__ erep,
                         const int* __restrict__ pair_idx,
                         u16* __restrict__ prod) {
  int r  = blockIdx.x;
  int i0 = pair_idx[2 * r];
  int i1 = pair_idx[2 * r + 1];
  const uint2* s0 = (const uint2*)(erep + (size_t)i0 * 1024);        // head: cols 0..511
  const uint2* s1 = (const uint2*)(erep + (size_t)i1 * 1024 + 512);  // tail: cols 512..1023
  uint2* d = (uint2*)(prod + (size_t)r * 1024);
  int t = threadIdx.x;                       // 256 threads, 256 x 8B chunks
  d[t] = (t < 128) ? s0[t] : s1[t - 128];
}

// ---------------------------------------------------------------------------
// Kernel 5: fused  GEMM2 -> +b_post_cat -> x union -> GEMM3 (+GEMM4) -> epilogue
// Workgroup = 128 rel-rows, 8 wave32s. Wave (a2,b2) owns a 2x2 grid of 16x16 tiles.
// LDS: 256KB prod tile + 2x16KB double-buffered W_post_cat K-slice + 16KB V
// stage = 304KB. Staging runs on the Tensor Data Mover (TENSORcnt), with the
// next slice's DMA issued before computing the current one.
// ---------------------------------------------------------------------------
__global__ __launch_bounds__(256)
void k_fused(const u16*  __restrict__ prod,    // 65536x1024 bf16
             const u16*  __restrict__ wpcT,    // 4096x1024 bf16
             const u16*  __restrict__ wrelT,   // 64x4096 bf16 (padded)
             const u16*  __restrict__ wctxT,   // 64x1024 bf16 (padded)
             const float* __restrict__ unionF, // 65536x4096 f32
             const float* __restrict__ bpc,    // 4096
             const float* __restrict__ brel,   // 51
             const float* __restrict__ bctx,   // 51
             const int*  __restrict__ pair_idx,
             const int*  __restrict__ obj_preds,
             const float* __restrict__ freq,   // 151*151 x 51
             float* __restrict__ out) {        // 65536 x 51
  __shared__ __align__(16) u16 sP[128 * 1024];   // prod tile        (256 KB)
  __shared__ __align__(16) u16 sB[2][64 * 128];  // W_post_cat slice (2x16 KB)
  __shared__ __align__(16) u16 sV[128 * 64];     // visual_rep stage (16 KB)

  const int tid = threadIdx.x;
  const int w   = tid >> 5;
  const int ln  = tid & 31;
  const int lh  = ln >> 4, lc = ln & 15;
  const int a2  = w & 3;                       // wave's M-pair (0..3)
  const int b2  = w >> 2;                      // wave's N-pair (0..1)
  const long m0 = (long)blockIdx.x * 128;

  // ---- stage the 128x1024 bf16 prod tile into LDS (read once per block) ----
#if HAVE_TDM
  // each wave DMAs its own 16x1024 bf16 tile (32 KB) via the TDM
  tdm_load_2d((u32)(uintptr_t)(sP + (size_t)w * 16 * 1024),
              prod + (size_t)(m0 + w * 16) * 1024,
              /*td0=*/1024, /*td1=*/16, /*tile0=*/1024, /*tile1=*/16,
              /*stride0=*/1024);
  if (w == 0)   // prefetch first W_post_cat slice (nch=0, kb=0) into sB[0]
    tdm_load_2d((u32)(uintptr_t)&sB[0][0], wpcT,
                /*td0=*/128, /*td1=*/64, /*tile0=*/128, /*tile1=*/64,
                /*stride0=*/1024);
  wait_tensor0();
  __syncthreads();
#else
  {
    const uint4* src = (const uint4*)(prod + (size_t)m0 * 1024);
    uint4* dst = (uint4*)sP;
    for (int i = tid; i < (128 * 1024) / 8; i += 256) dst[i] = src[i];
  }
  __syncthreads();
#endif

  v8f acc[2][2] = {};                          // final 128x64 block accumulators

  // ================= loop over POOL dimension in 64-col chunks ==============
  for (int nch = 0; nch < POOL / 64; ++nch) {
    v8f g[2][2] = {};                          // ctx_gate tile accumulators

    // ---- GEMM2: K=1024 in 8 slices of 128; B slice staged by the TDM and
    //      double-buffered so DMA of slice s+1 overlaps WMMA on slice s ----
    for (int kb = 0; kb < 8; ++kb) {
      const int s = nch * 8 + kb;              // global slice index (0..511)
#if HAVE_TDM
      if (w == 0) wait_tensor0();              // slice s landed in sB[s&1]
      __syncthreads();
      if (w == 0 && s + 1 < (POOL / 64) * 8) { // kick DMA for slice s+1
        int s1 = s + 1;
        int nch1 = s1 >> 3, kb1 = s1 & 7;
        tdm_load_2d((u32)(uintptr_t)&sB[s1 & 1][0],
                    wpcT + (size_t)(nch1 * 64) * 1024 + kb1 * 128,
                    128, 64, 128, 64, 1024);
      }
#else
      __syncthreads();                         // previous users of sB done
      {
        const u16* srcW = wpcT + (size_t)(nch * 64) * 1024 + kb * 128;
        for (int u = tid; u < 1024; u += 256) { // 1024 x 16B = 16 KB
          int e  = u * 8;
          int nl = e >> 7, kl = e & 127;
          *(uint4*)(&sB[s & 1][0] + nl * 128 + kl) =
              *(const uint4*)(srcW + (size_t)nl * 1024 + kl);
        }
      }
      __syncthreads();
#endif
      const u16* sBc = &sB[s & 1][0];
#pragma unroll
      for (int ks = 0; ks < 4; ++ks) {
        int kglob = kb * 128 + ks * 32;
        Frag A[2], B[2];
#pragma unroll
        for (int i = 0; i < 2; ++i) {
          int row = (2 * a2 + i) * 16 + lc;
          const uint4* ap = (const uint4*)(sP + row * 1024 + kglob + 16 * lh);
          A[i].q[0] = ap[0];
          A[i].q[1] = ap[1];
        }
#pragma unroll
        for (int i = 0; i < 2; ++i) {
          int ncl = (2 * b2 + i) * 16 + lc;
          const uint4* bp = (const uint4*)(sBc + ncl * 128 + ks * 32 + 16 * lh);
          B[i].q[0] = bp[0];
          B[i].q[1] = bp[1];
        }
        g[0][0] = wmma_bf16(A[0], B[0], g[0][0]);
        g[0][1] = wmma_bf16(A[0], B[1], g[0][1]);
        g[1][0] = wmma_bf16(A[1], B[0], g[1][0]);
        g[1][1] = wmma_bf16(A[1], B[1], g[1][1]);
      }
    }

    // ---- +b_post_cat, multiply by streamed union_features, stage V (bf16) ----
#pragma unroll
    for (int i = 0; i < 2; ++i) {
      int rloc = (2 * a2 + i) * 16;
#pragma unroll
      for (int jn = 0; jn < 2; ++jn) {
        int ncl = (2 * b2 + jn) * 16 + lc;     // 0..63
        int ncg = nch * 64 + ncl;              // 0..4095
        float bias = bpc[ncg];
#pragma unroll
        for (int j = 0; j < 8; ++j) {
          long rg = m0 + rloc + j + 8 * lh;
          const float* uptr = unionF + (size_t)rg * POOL + ncg;
          float u = *uptr;
          if (nch + 1 < POOL / 64)
            __builtin_prefetch(uptr + 64, 0, 1);  // next chunk's U line
          sV[(rloc + j + 8 * lh) * 64 + ncl] = f2bf((g[i][jn][j] + bias) * u);
        }
      }
    }
    __syncthreads();

    // ---- GEMM3: ACC += V(128x64) @ W_rel[nch*64 .. +64), :]  (K=64) ----
#pragma unroll
    for (int ks = 0; ks < 2; ++ks) {
      Frag A[2], B[2];
#pragma unroll
      for (int i = 0; i < 2; ++i) {
        int row = (2 * a2 + i) * 16 + lc;
        const uint4* ap = (const uint4*)(sV + row * 64 + ks * 32 + 16 * lh);
        A[i].q[0] = ap[0];
        A[i].q[1] = ap[1];
      }
#pragma unroll
      for (int i = 0; i < 2; ++i) {
        int colW = (2 * b2 + i) * 16 + lc;     // 0..63 (51 real + pad)
        const uint4* bp = (const uint4*)(wrelT + (size_t)colW * POOL +
                                         nch * 64 + ks * 32 + 16 * lh);
        B[i].q[0] = bp[0];
        B[i].q[1] = bp[1];
      }
      acc[0][0] = wmma_bf16(A[0], B[0], acc[0][0]);
      acc[0][1] = wmma_bf16(A[0], B[1], acc[0][1]);
      acc[1][0] = wmma_bf16(A[1], B[0], acc[1][0]);
      acc[1][1] = wmma_bf16(A[1], B[1], acc[1][1]);
    }
  }

  // ---- GEMM4: ACC += prod(128x1024) @ W_ctx (K=1024), A from LDS tile ----
  for (int k = 0; k < 1024; k += 32) {
    Frag A[2], B[2];
#pragma unroll
    for (int i = 0; i < 2; ++i) {
      int row = (2 * a2 + i) * 16 + lc;
      const uint4* ap = (const uint4*)(sP + row * 1024 + k + 16 * lh);
      A[i].q[0] = ap[0];
      A[i].q[1] = ap[1];
    }
#pragma unroll
    for (int i = 0; i < 2; ++i) {
      int colW = (2 * b2 + i) * 16 + lc;
      const uint4* bp = (const uint4*)(wctxT + (size_t)colW * 1024 + k + 16 * lh);
      B[i].q[0] = bp[0];
      B[i].q[1] = bp[1];
    }
    acc[0][0] = wmma_bf16(A[0], B[0], acc[0][0]);
    acc[0][1] = wmma_bf16(A[0], B[1], acc[0][1]);
    acc[1][0] = wmma_bf16(A[1], B[0], acc[1][0]);
    acc[1][1] = wmma_bf16(A[1], B[1], acc[1][1]);
  }

  // ---- epilogue: +b_rel +b_ctx +freq_table[pair_pred], store f32 ----
#pragma unroll
  for (int i = 0; i < 2; ++i) {
#pragma unroll
    for (int jn = 0; jn < 2; ++jn) {
      int colW = (2 * b2 + jn) * 16 + lc;
      if (colW < NRC) {
        float bb = brel[colW] + bctx[colW];
#pragma unroll
        for (int j = 0; j < 8; ++j) {
          long rg = m0 + (2 * a2 + i) * 16 + j + 8 * lh;
          int p0 = pair_idx[2 * rg];
          int p1 = pair_idx[2 * rg + 1];
          size_t pp = (size_t)obj_preds[p0] * NOC + obj_preds[p1];
          out[(size_t)rg * NRC + colW] =
              acc[i][jn][j] + bb + freq[pp * NRC + colW];
        }
      }
    }
  }
}

// ---------------------------------------------------------------------------
// Host-side launcher
// ---------------------------------------------------------------------------
extern "C" void kernel_launch(void* const* d_in, const int* in_sizes, int n_in,
                              void* d_out, int out_size, void* d_ws, size_t ws_size,
                              hipStream_t stream) {
  (void)in_sizes; (void)n_in; (void)out_size; (void)ws_size;

  const float* edge_ctx   = (const float*)d_in[0];
  const int*   obj_preds  = (const int*)  d_in[1];
  const int*   pair_idx   = (const int*)  d_in[2];
  const float* unionF     = (const float*)d_in[3];
  const float* W_post_emb = (const float*)d_in[4];
  const float* b_post_emb = (const float*)d_in[5];
  const float* W_post_cat = (const float*)d_in[6];
  const float* b_post_cat = (const float*)d_in[7];
  const float* W_rel      = (const float*)d_in[8];
  const float* b_rel      = (const float*)d_in[9];
  const float* W_ctx      = (const float*)d_in[10];
  const float* b_ctx      = (const float*)d_in[11];
  const float* freq       = (const float*)d_in[12];
  float* out = (float*)d_out;

  // Workspace layout (bf16 buffers), ~170 MB total
  u16* ws = (u16*)d_ws;
  size_t off = 0;
  u16* ecb   = ws + off; off += (size_t)N_OBJ * HID;        // edge_ctx bf16
  u16* wpeT  = ws + off; off += (size_t)1024 * HID;         // W_post_emb^T
  u16* erep  = ws + off; off += (size_t)N_OBJ * 1024;       // edge_rep bf16
  u16* wpcT  = ws + off; off += (size_t)POOL * 1024;        // W_post_cat^T
  u16* wrelT = ws + off; off += (size_t)NPAD * POOL;        // W_rel^T padded
  u16* wctxT = ws + off; off += (size_t)NPAD * 1024;        // W_ctx^T padded
  u16* prod  = ws + off; off += (size_t)N_REL * 1024;       // prod_rep bf16

  // 1) conversions / transposes (all tiny vs main GEMM)
  {
    int n = N_OBJ * HID;
    k_cvt<<<(n + 255) / 256, 256, 0, stream>>>(edge_ctx, ecb, n);
  }
  { int t = 1024 * HID;  k_tcvt<<<(t + 255) / 256, 256, 0, stream>>>(W_post_emb, wpeT, HID, 1024, 1024); }
  { int t = POOL * 1024; k_tcvt<<<(t + 255) / 256, 256, 0, stream>>>(W_post_cat, wpcT, 1024, POOL, POOL); }
  { int t = NPAD * POOL; k_tcvt<<<(t + 255) / 256, 256, 0, stream>>>(W_rel, wrelT, POOL, NRC, NPAD); }
  { int t = NPAD * 1024; k_tcvt<<<(t + 255) / 256, 256, 0, stream>>>(W_ctx, wctxT, 1024, NRC, NPAD); }

  // 2) GEMM1: edge_rep (8192 waves -> 1024 blocks of 8 wave32s)
  k_gemm1<<<1024, 256, 0, stream>>>(ecb, wpeT, b_post_emb, erep);

  // 3) gather prod_rep
  k_gather<<<N_REL, 256, 0, stream>>>(erep, pair_idx, prod);

  // 4) fused GEMM2 * union -> GEMM3 + GEMM4 + epilogue
  k_fused<<<N_REL / 128, 256, 0, stream>>>(prod, wpcT, wrelT, wctxT, unionF,
                                           b_post_cat, b_rel, b_ctx,
                                           pair_idx, obj_preds, freq, out);
}
